// RNN_Attention_7430293422159
// MI455X (gfx1250) — compile-verified, matrix operand order checked by reading, not run on hardware
//
#include <hip/hip_runtime.h>

#define SEQ   2048
#define BATCH 256
#define IN    32
#define HID   128

typedef __attribute__((ext_vector_type(16))) _Float16 v16h;
typedef __attribute__((ext_vector_type(8)))  float    v8f;

union V16H { v16h v; _Float16 h[16]; int4 q[2]; };
union H8   { int4 q; _Float16 h[8]; };

__device__ __forceinline__ float fast_tanh(float x) {
    // tanh(x) = 1 - 2/(exp(2x)+1); exact at +-inf, v_exp + v_rcp fast path
    float e = __expf(2.0f * x);
    return 1.0f - 2.0f * __builtin_amdgcn_rcpf(e + 1.0f);
}

// ---------------------------------------------------------------------------
// Phase 1: sequential recurrence. 16 workgroups (one per 16-row batch tile),
// 8 waves each; wave w owns output columns [16w,16w+16). Wi/Wh slices live in
// VGPRs as WMMA-B f16 operands for the whole kernel; h0 state is double-
// buffered in LDS (f16, WMMA-A source) -> ONE barrier per step. hstart/ht/h0
// per-element copies live in registers (each wave always owns the same
// (row,col) elements). Two WMMA accumulator chains to cut dependent depth.
// ---------------------------------------------------------------------------
__global__ __launch_bounds__(256)
void rnn_recur_7430293422159(const float* __restrict__ x,
                             const float* __restrict__ Wi, const float* __restrict__ bi,
                             const float* __restrict__ Wh, const float* __restrict__ bh,
                             const int*   __restrict__ resD,
                             _Float16*    __restrict__ hid /* (S,B,H) f16 */) {
    __shared__ _Float16 h0_lds[2][16][144];   // double-buffered h0 state
    __shared__ _Float16 hn_lds[2][16][128];   // double-buffered store staging
    __shared__ _Float16 x_lds[2][16][40];     // double-buffered x_t tile

    const int t      = threadIdx.x;
    const int lane   = t & 31;
    const int w      = t >> 5;                // wave id 0..7
    const int nLane  = lane & 15;             // N index / M row for A loads
    const int hiHalf = lane >> 4;             // lane group 0/1 (K-offset +8)
    const int col    = w * 16 + nLane;        // output column in H
    const int b0     = blockIdx.x * 16;       // batch tile
    const int rd     = resD[0];

    // ---- resident B operands: Wi[:,col-slice], Wh[:,col-slice] as f16 ----
    v16h bWi, bWh[4];
    {
        V16H u;
        #pragma unroll
        for (int e = 0; e < 16; ++e) {
            int k = hiHalf * 8 + (e & 7) + ((e >> 3) << 4);     // K in [0,32)
            u.h[e] = (_Float16)Wi[k * HID + col];
        }
        bWi = u.v;
        #pragma unroll
        for (int kb = 0; kb < 4; ++kb) {
            V16H u2;
            #pragma unroll
            for (int e = 0; e < 16; ++e) {
                int k = kb * 32 + hiHalf * 8 + (e & 7) + ((e >> 3) << 4);
                u2.h[e] = (_Float16)Wh[k * HID + col];
            }
            bWh[kb] = u2.v;
        }
    }
    const float biasLane = bi[col] + bh[col];

    // ---- init: zero h0[0], stage x[0] into buffer 0 ----
    for (int i = t; i < 16 * 144; i += 256) ((_Float16*)h0_lds[0])[i] = (_Float16)0.0f;
    const int xrow = t >> 3, xqc = t & 7;
    if (t < 128) {
        float4 f = *(const float4*)(x + ((size_t)(b0 + xrow) * SEQ + 0) * IN + xqc * 4);
        _Float16* p = &x_lds[0][xrow][xqc * 4];
        p[0] = (_Float16)f.x; p[1] = (_Float16)f.y; p[2] = (_Float16)f.z; p[3] = (_Float16)f.w;
    }
    __syncthreads();

    float h0_r[8], ht_r[8], hs_r[8];
    #pragma unroll
    for (int v = 0; v < 8; ++v) { h0_r[v] = 0.f; ht_r[v] = 0.f; hs_r[v] = 0.f; }

    int cnt = 0;   // running s % rd, maintained incrementally (no div in loop)

    for (int s = 0; s < SEQ; ++s) {
        const int p = s & 1;

        // prefetch next x tile into registers (overlaps WMMA)
        float4 nxt = make_float4(0.f, 0.f, 0.f, 0.f);
        if (t < 128 && s + 1 < SEQ) {
            nxt = *(const float4*)(x + ((size_t)(b0 + xrow) * SEQ + (s + 1)) * IN + xqc * 4);
            if (s + 8 < SEQ)
                __builtin_prefetch(x + ((size_t)(b0 + xrow) * SEQ + (s + 8)) * IN + xqc * 4, 0, 0);
        }

        // A operands from LDS (documented 16-bit A layout; 16B ds_load chunks)
        V16H ax;
        {
            const _Float16* xr = &x_lds[p][nLane][hiHalf * 8];
            ax.q[0] = *(const int4*)(xr);
            ax.q[1] = *(const int4*)(xr + 16);
        }
        V16H ah[4];
        const _Float16* h0row = &h0_lds[p][nLane][0];
        #pragma unroll
        for (int kb = 0; kb < 4; ++kb) {
            const _Float16* hr = h0row + kb * 32 + hiHalf * 8;
            ah[kb].q[0] = *(const int4*)(hr);
            ah[kb].q[1] = *(const int4*)(hr + 16);
        }

        // hn tile = x_t*Wi + h0*Wh : two independent accumulation chains
        v8f c0 = {}, c1 = {};
        c0 = __builtin_amdgcn_wmma_f32_16x16x32_f16(false, ax.v,     false, bWi,    (short)0, c0, false, false);
        c1 = __builtin_amdgcn_wmma_f32_16x16x32_f16(false, ah[0].v,  false, bWh[0], (short)0, c1, false, false);
        c0 = __builtin_amdgcn_wmma_f32_16x16x32_f16(false, ah[1].v,  false, bWh[1], (short)0, c0, false, false);
        c1 = __builtin_amdgcn_wmma_f32_16x16x32_f16(false, ah[2].v,  false, bWh[2], (short)0, c1, false, false);
        c0 = __builtin_amdgcn_wmma_f32_16x16x32_f16(false, ah[3].v,  false, bWh[3], (short)0, c0, false, false);

        // exclusive branch id, same priority as reference _branch_ids
        const int bid = (s == 0) ? 0 : (s == SEQ - 2) ? 1 : (rd == 1) ? 4
                      : (cnt == 0) ? 2 : 3;
        #pragma unroll
        for (int v = 0; v < 8; ++v) {
            const int M = v + hiHalf * 8;
            const float hn = c0[v] + c1[v] + biasLane;
            if (bid == 0) {
                hs_r[v] = hn;                      // h0/ht unchanged (zero)
            } else {
                float pre = hn;
                if      (bid == 1) pre = hn + hs_r[v];
                else if (bid == 4) pre = hn + h0_r[v];
                else if (bid == 2) pre = hn + ht_r[v];
                float cand = fast_tanh(pre);
                if (bid == 2) ht_r[v] = hn;
                h0_r[v] = cand;
            }
            h0_lds[p ^ 1][M][col] = (_Float16)h0_r[v];   // next-step state
            hn_lds[p][M][col]     = (_Float16)hn;        // this-step output
        }
        if (t < 128 && s + 1 < SEQ) {                    // stage next x tile
            _Float16* xp = &x_lds[p ^ 1][xrow][xqc * 4];
            xp[0] = (_Float16)nxt.x; xp[1] = (_Float16)nxt.y;
            xp[2] = (_Float16)nxt.z; xp[3] = (_Float16)nxt.w;
        }
        __syncthreads();   // single barrier: publishes h0[p^1]/hn[p]/x[p^1]

        // stream hn tile to global hiddens: 4KB, one b128 per thread, coalesced
        int4* dst = (int4*)(hid + (size_t)s * (BATCH * HID) + (size_t)b0 * HID);
        dst[t] = ((const int4*)hn_lds[p])[t];

        cnt = (cnt + 1 == rd) ? 0 : cnt + 1;
    }
}

// ---------------------------------------------------------------------------
// Phase 1b: scores[s*B+b] = hid[s,b,:].Wa + ba  (bandwidth-bound GEMV)
// ---------------------------------------------------------------------------
__global__ __launch_bounds__(256)
void rnn_score_7430293422159(const _Float16* __restrict__ hid,
                             const float* __restrict__ Wa, const float* __restrict__ ba,
                             float* __restrict__ scores) {
    __shared__ float wa_s[HID];
    const int t = threadIdx.x;
    if (t < HID) wa_s[t] = Wa[t];
    __syncthreads();
    const size_t idx = (size_t)blockIdx.x * 256 + t;       // flat (s*B + b)
    const _Float16* hp = hid + idx * HID;
    float d = ba[0];
    #pragma unroll 4
    for (int h = 0; h < HID; h += 8) {
        H8 u; u.q = *(const int4*)(hp + h);
        d += (float)u.h[0] * wa_s[h + 0] + (float)u.h[1] * wa_s[h + 1]
           + (float)u.h[2] * wa_s[h + 2] + (float)u.h[3] * wa_s[h + 3]
           + (float)u.h[4] * wa_s[h + 4] + (float)u.h[5] * wa_s[h + 5]
           + (float)u.h[6] * wa_s[h + 6] + (float)u.h[7] * wa_s[h + 7];
    }
    scores[idx] = d;
}

// ---------------------------------------------------------------------------
// Phase 2: bug-faithful softmax row r over scores[r*2048..), then
// out[r] = sum_j att_j * <hid[r*262144 + j*128 + :], Wd[j*128 + :]> + bd
// (Wd is 1MB -> L2-resident across the 256 blocks; hid streamed once.)
// ---------------------------------------------------------------------------
__global__ __launch_bounds__(256)
void rnn_out_7430293422159(const float* __restrict__ scores, const _Float16* __restrict__ hid,
                           const float* __restrict__ Wd, const float* __restrict__ bd,
                           float* __restrict__ out) {
    __shared__ float red[256];
    const int r = blockIdx.x, t = threadIdx.x;
    const float* sc = scores + (size_t)r * 2048;

    float m = -3.4e38f;
    for (int j = t; j < 2048; j += 256) m = fmaxf(m, sc[j]);
    red[t] = m; __syncthreads();
    for (int o = 128; o > 0; o >>= 1) { if (t < o) red[t] = fmaxf(red[t], red[t + o]); __syncthreads(); }
    m = red[0]; __syncthreads();

    float z = 0.f;
    for (int j = t; j < 2048; j += 256) z += __expf(sc[j] - m);
    red[t] = z; __syncthreads();
    for (int o = 128; o > 0; o >>= 1) { if (t < o) red[t] += red[t + o]; __syncthreads(); }
    const float invZ = __builtin_amdgcn_rcpf(red[0]); __syncthreads();

    float acc = 0.f;
    for (int j = t; j < 2048; j += 256) {
        const float a = __expf(sc[j] - m) * invZ;
        const _Float16* hp = hid + (size_t)r * 262144 + (size_t)j * 128;
        const float*    wp = Wd + (size_t)j * 128;
        float d = 0.f;
        #pragma unroll 2
        for (int h = 0; h < 128; h += 8) {
            H8 u; u.q = *(const int4*)(hp + h);
            float4 w0 = *(const float4*)(wp + h);
            float4 w1 = *(const float4*)(wp + h + 4);
            d += (float)u.h[0] * w0.x + (float)u.h[1] * w0.y
               + (float)u.h[2] * w0.z + (float)u.h[3] * w0.w
               + (float)u.h[4] * w1.x + (float)u.h[5] * w1.y
               + (float)u.h[6] * w1.z + (float)u.h[7] * w1.w;
        }
        acc += a * d;
    }
    red[t] = acc; __syncthreads();
    for (int o = 128; o > 0; o >>= 1) { if (t < o) red[t] += red[t + o]; __syncthreads(); }
    if (t == 0) out[r] = red[0] + bd[0];
}

extern "C" void kernel_launch(void* const* d_in, const int* in_sizes, int n_in,
                              void* d_out, int out_size, void* d_ws, size_t ws_size,
                              hipStream_t stream) {
    (void)in_sizes; (void)n_in; (void)out_size; (void)ws_size;
    const float* x  = (const float*)d_in[0];
    const float* Wi = (const float*)d_in[1];
    const float* bi = (const float*)d_in[2];
    const float* Wh = (const float*)d_in[3];
    const float* bh = (const float*)d_in[4];
    const float* Wa = (const float*)d_in[5];
    const float* ba = (const float*)d_in[6];
    const float* Wd = (const float*)d_in[7];
    const float* bd = (const float*)d_in[8];
    const int*   rd = (const int*)d_in[9];
    float* out = (float*)d_out;

    char* ws = (char*)d_ws;
    _Float16* hid   = (_Float16*)ws;                                    // 128 MB
    float*    score = (float*)(ws + (size_t)SEQ * BATCH * HID * 2);     // 2 MB

    rnn_recur_7430293422159<<<BATCH / 16, 256, 0, stream>>>(x, Wi, bi, Wh, bh, rd, hid);
    rnn_score_7430293422159<<<(SEQ * BATCH) / 256, 256, 0, stream>>>(hid, Wa, ba, score);
    rnn_out_7430293422159<<<BATCH, 256, 0, stream>>>(score, hid, Wd, bd, out);
}
